// JetFormer_5634997092660
// MI455X (gfx1250) — matrix operand order repaired and checked by reading, not compiled, not run on hardware
//
#include <hip/hip_runtime.h>
#include <hip/hip_bf16.h>

// ---------------- model constants ----------------
#define BB   2
#define TT   256
#define DD   768
#define HH   12
#define HDIM 64
#define FF   3072
#define LL   12
#define VV   32000
#define NN   256      // image tokens
#define DTOKN 192
#define DAR  128
#define KMIX 128
#define SS   514      // 1 + T + 1 + N
#define SLD  520      // padded row stride for the score matrix (16B-aligned rows)

typedef __bf16 bf16;
typedef __attribute__((ext_vector_type(16))) __bf16 v16bf;
typedef __attribute__((ext_vector_type(2)))  __bf16 v2bf;
typedef __attribute__((ext_vector_type(8)))  float  v8f;
typedef __attribute__((ext_vector_type(2)))  unsigned v2u;

// two fp32 -> packed bf16x2 via native converts (RNE fptrunc -> v_cvt*)
__device__ __forceinline__ unsigned pack2bf(float a, float b) {
  v2bf v = { (__bf16)a, (__bf16)b };
  return __builtin_bit_cast(unsigned, v);
}

__device__ __forceinline__ float gelu_tanh(float x) {
  return 0.5f * x * (1.f + tanhf(0.7978845608028654f * (x + 0.044715f * x * x * x)));
}

// -----------------------------------------------------------------------------
// Batched GEMM: C[z] (M x N) = A[z] (M x K) * B[z] (K x N) [+bias][+acc][gelu]
// fp32 in/out, bf16 WMMA core (v_wmma_f32_16x16x32_bf16).
// Block = 256 thr = 8 waves; block tile 128x64; wave quadrant 32x32 = 2x2 WMMA.
// Software pipeline: next K-slab loaded into registers while WMMAs run on the
// current LDS buffer; double-buffered LDS, ONE barrier per K step.
// LDS tiles staged in fragment order -> each lane reads its whole fragment as
// one aligned 32B ds read (ds_load_b128 x2).
//   A frag map: (row,kk) -> laneGrp=kk>>4,      i=kk&15
//   B frag map: (k,n)    -> laneGrp=k>>4,      i=k&15
// A and transB-B fast paths use float4 global loads + ds_store_b64 (a float4's
// 4 K-elements map to one lane at consecutive i). Requires lda/ldb % 4 == 0 and
// 16B-aligned bases -- true for every call site (scores use padded SLD=520).
// Edge tiles use clamped scalar loads + zero-select (no divergent branches).
// z decomposed as (batch, head): base += b*s?b + h*s?h with h = z % zdimH.
// transB: B stored row-major (N x K): logical B[k][n] = Bm[n*ldb + k].
// -----------------------------------------------------------------------------
__global__ __launch_bounds__(256)
void gemm_wmma(const float* __restrict__ A, int lda, long sAb, long sAh,
               const float* __restrict__ Bm, int ldb, long sBb, long sBh, int transB,
               float* __restrict__ C, int ldc, long sCb, long sCh,
               const float* __restrict__ bias,
               int M, int N, int Kd, int zdimH, int accumulate, int epilogue)
{
  int z  = blockIdx.z;
  int bz = z / zdimH, hz = z % zdimH;
  A  += (long)bz * sAb + (long)hz * sAh;
  Bm += (long)bz * sBb + (long)hz * sBh;
  C  += (long)bz * sCb + (long)hz * sCh;

  __shared__ __align__(32) bf16 Af[2][8][32][16];  // [buf][rowHalf][lane][i]
  __shared__ __align__(32) bf16 Bf[2][4][32][16];  // [buf][colHalf][lane][i]

  const int m0 = blockIdx.y * 128, n0 = blockIdx.x * 64;
  const int tid  = threadIdx.x;
  const int lane = tid & 31;
  const int wv = tid >> 5;
  const int wr = wv >> 1, wc = wv & 1;      // wave quadrant: rows wr*32.., cols wc*32..
  const int lm = lane & 15, lg = lane >> 4;
  const bool aIn = (m0 + 128 <= M);
  const bool bIn = (n0 + 64 <= N);

  v8f acc[2][2] = {};
  float4 ra[4];    // staged A slab: 128x32 = 1024 float4 / 256 thr = 4 each
  float4 rbT[2];   // staged B slab (transB):  32x64 = 512 float4 / 256 thr = 2
  float2 rbN[4];   // staged B slab (normal):  32x64 = 1024 float2 / 256 thr = 4

  auto loadA = [&](int k0) {
    const bool fast = aIn && (k0 + 32 <= Kd);
    #pragma unroll
    for (int r = 0; r < 4; ++r) {
      int id = tid + r * 256;
      int row = id >> 3, kq = (id & 7) * 4;
      if (fast) {
        ra[r] = *(const float4*)(A + (m0 + row) * lda + k0 + kq);
      } else {                            // clamped scalar loads + zero-select
        int ar = m0 + row;
        int arl = ar < M ? ar : M - 1;
        float vv[4];
        #pragma unroll
        for (int e = 0; e < 4; ++e) {
          int ac = k0 + kq + e;
          int acl = ac < Kd ? ac : Kd - 1;
          float v = A[arl * lda + acl];
          vv[e] = (ar < M && ac < Kd) ? v : 0.f;
        }
        ra[r] = make_float4(vv[0], vv[1], vv[2], vv[3]);
      }
    }
  };

  auto loadB = [&](int k0) {
    const bool fast = bIn && (k0 + 32 <= Kd);
    if (transB) {
      #pragma unroll
      for (int r = 0; r < 2; ++r) {
        int id = tid + r * 256;
        int nc = id >> 3, kq = (id & 7) * 4;
        if (fast) {
          rbT[r] = *(const float4*)(Bm + (n0 + nc) * ldb + k0 + kq);
        } else {
          int bc = n0 + nc;
          int bcl = bc < N ? bc : N - 1;
          float vv[4];
          #pragma unroll
          for (int e = 0; e < 4; ++e) {
            int bk = k0 + kq + e;
            int bkl = bk < Kd ? bk : Kd - 1;
            float v = Bm[bcl * ldb + bkl];
            vv[e] = (bc < N && bk < Kd) ? v : 0.f;
          }
          rbT[r] = make_float4(vv[0], vv[1], vv[2], vv[3]);
        }
      }
    } else {
      #pragma unroll
      for (int r = 0; r < 4; ++r) {
        int id = tid + r * 256;
        int nc = id & 63, kk = (id >> 6) * 2;
        if (fast) {
          rbN[r].x = Bm[(k0 + kk    ) * ldb + n0 + nc];
          rbN[r].y = Bm[(k0 + kk + 1) * ldb + n0 + nc];
        } else {
          int bc = n0 + nc;
          int bcl = bc < N ? bc : N - 1;
          int k1 = k0 + kk, k2 = k0 + kk + 1;
          int k1l = k1 < Kd ? k1 : Kd - 1;
          int k2l = k2 < Kd ? k2 : Kd - 1;
          float v0 = Bm[k1l * ldb + bcl];
          float v1 = Bm[k2l * ldb + bcl];
          rbN[r].x = (bc < N && k1 < Kd) ? v0 : 0.f;
          rbN[r].y = (bc < N && k2 < Kd) ? v1 : 0.f;
        }
      }
    }
  };

  auto storeTiles = [&](int cur) {
    #pragma unroll
    for (int r = 0; r < 4; ++r) {        // A: float4 -> one ds_store_b64
      int id = tid + r * 256;
      int row = id >> 3, kq = (id & 7) * 4;
      int l = ((kq >> 4) << 4) | (row & 15);
      int i = kq & 15;
      v2u w;
      w.x = pack2bf(ra[r].x, ra[r].y);
      w.y = pack2bf(ra[r].z, ra[r].w);
      *(v2u*)&Af[cur][row >> 4][l][i] = w;
    }
    if (transB) {
      #pragma unroll
      for (int r = 0; r < 2; ++r) {      // B: float4 -> one ds_store_b64
        int id = tid + r * 256;
        int nc = id >> 3, kq = (id & 7) * 4;
        int l = ((kq >> 4) << 4) | (nc & 15);
        v2u w;
        w.x = pack2bf(rbT[r].x, rbT[r].y);
        w.y = pack2bf(rbT[r].z, rbT[r].w);
        *(v2u*)&Bf[cur][nc >> 4][l][kq & 15] = w;
      }
    } else {
      #pragma unroll
      for (int r = 0; r < 4; ++r) {
        int id = tid + r * 256;
        int nc = id & 63, kk = (id >> 6) * 2;
        int l = ((kk >> 4) << 4) | (nc & 15);
        *(unsigned*)&Bf[cur][nc >> 4][l][kk & 15] = pack2bf(rbN[r].x, rbN[r].y);
      }
    }
  };

  // ---- pipelined main loop: one barrier per K step ----
  loadA(0);
  loadB(0);
  int cur = 0;
  for (int k0 = 0; k0 < Kd; k0 += 32) {
    storeTiles(cur);
    __syncthreads();
    if (k0 + 32 < Kd) {       // issue next slab's global loads early
      loadA(k0 + 32);
      loadB(k0 + 32);
    }
    v16bf a0 = *(const v16bf*)&Af[cur][wr * 2 + 0][lane][0];
    v16bf a1 = *(const v16bf*)&Af[cur][wr * 2 + 1][lane][0];
    v16bf b0 = *(const v16bf*)&Bf[cur][wc * 2 + 0][lane][0];
    v16bf b1 = *(const v16bf*)&Bf[cur][wc * 2 + 1][lane][0];
    acc[0][0] = __builtin_amdgcn_wmma_f32_16x16x32_bf16(false, a0, false, b0, (short)0, acc[0][0], false, false);
    acc[0][1] = __builtin_amdgcn_wmma_f32_16x16x32_bf16(false, a0, false, b1, (short)0, acc[0][1], false, false);
    acc[1][0] = __builtin_amdgcn_wmma_f32_16x16x32_bf16(false, a1, false, b0, (short)0, acc[1][0], false, false);
    acc[1][1] = __builtin_amdgcn_wmma_f32_16x16x32_bf16(false, a1, false, b1, (short)0, acc[1][1], false, false);
    cur ^= 1;
  }

  // ---- epilogue / store (C/D layout: VGPR j -> M = j + 8*lg; N = lane&15) ----
  const bool interior = aIn && bIn;
  #pragma unroll
  for (int ii = 0; ii < 2; ++ii) {
    #pragma unroll
    for (int jj = 0; jj < 2; ++jj) {
      int col = n0 + wc * 32 + jj * 16 + lm;
      int rbase = m0 + wr * 32 + ii * 16 + 8 * lg;
      if (interior) {
        float bv = bias ? bias[col] : 0.f;
        #pragma unroll
        for (int j = 0; j < 8; ++j) {
          long o = (long)(rbase + j) * ldc + col;
          float v = acc[ii][jj][j] + bv;
          if (epilogue == 1) v = gelu_tanh(v);
          if (accumulate) v += C[o];
          C[o] = v;
        }
      } else {
        float bv = (bias && col < N) ? bias[col] : 0.f;
        #pragma unroll
        for (int j = 0; j < 8; ++j) {
          int row = rbase + j;
          if (row < M && col < N) {
            long o = (long)row * ldc + col;
            float v = acc[ii][jj][j] + bv;
            if (epilogue == 1) v = gelu_tanh(v);
            if (accumulate) v += C[o];
            C[o] = v;
          }
        }
      }
    }
  }
}

// ---------------- RMSNorm: one block (256 thr) per row of length D=768 -------
__global__ __launch_bounds__(256)
void rmsnorm_k(const float* __restrict__ in, const float* __restrict__ w,
               float* __restrict__ out)
{
  long row = blockIdx.x;
  const float* xp = in + row * DD;
  float* yp = out + row * DD;
  __shared__ float red[256];
  int t = threadIdx.x;
  float s = 0.f;
  for (int d = t; d < DD; d += 256) { float v = xp[d]; s += v * v; }
  red[t] = s; __syncthreads();
  for (int o = 128; o > 0; o >>= 1) { if (t < o) red[t] += red[t + o]; __syncthreads(); }
  float r = rsqrtf(red[0] / (float)DD + 1e-6f);
  for (int d = t; d < DD; d += 256) yp[d] = xp[d] * r * w[d];
}

// ---------------- RoPE (+optional q scale); x shape (B,S,heads,64) -----------
__global__ void rope_k(float* __restrict__ x, int heads, float scale)
{
  int idx = blockIdx.x * blockDim.x + threadIdx.x;
  int total = BB * SS * heads * 32;
  if (idx >= total) return;
  int j = idx & 31;
  int t = idx >> 5;                   // (b*S+s)*heads + h
  int s = (t / heads) % SS;
  long base = (long)t * 64;
  float freq = expf(-(float)j * (9.210340371976184f / 32.f));  // 10000^(-j/32)
  float ang = (float)s * freq;
  float c = cosf(ang), sn = sinf(ang);
  float x1 = x[base + j] * scale, x2 = x[base + j + 32] * scale;
  x[base + j]      = x1 * c - x2 * sn;
  x[base + j + 32] = x2 * c + x1 * sn;
}

// ---- softcap + causal mask + softmax over rows (row stride SLD, len SS) -----
__global__ __launch_bounds__(256)
void softmax_k(float* __restrict__ sc)
{
  long r = blockIdx.x;                // (b*H + h)*S + qrow
  int qrow = (int)(r % SS);
  float* p = sc + r * SLD;
  __shared__ float red[256];
  int t = threadIdx.x;
  float vals[3];
  float m = -1e30f;
  for (int c = t, i = 0; c < SS; c += 256, ++i) {
    float v = 50.f * tanhf(p[c] * 0.02f);       // CAP * tanh(s/CAP)
    vals[i] = v;
    if (c <= qrow) m = fmaxf(m, v);
  }
  red[t] = m; __syncthreads();
  for (int o = 128; o > 0; o >>= 1) { if (t < o) red[t] = fmaxf(red[t], red[t + o]); __syncthreads(); }
  m = red[0]; __syncthreads();
  float s = 0.f;
  for (int c = t, i = 0; c < SS; c += 256, ++i) {
    float e = (c <= qrow) ? expf(vals[i] - m) : 0.f;
    vals[i] = e; s += e;
  }
  red[t] = s; __syncthreads();
  for (int o = 128; o > 0; o >>= 1) { if (t < o) red[t] += red[t + o]; __syncthreads(); }
  float inv = 1.f / red[0];
  for (int c = t, i = 0; c < SS; c += 256, ++i) p[c] = vals[i] * inv;
}

// ---------------- patchify: (B,128,128,3) -> (B,N,192) -----------------------
__global__ void patchify_k(const float* __restrict__ img, float* __restrict__ p)
{
  int idx = blockIdx.x * blockDim.x + threadIdx.x;
  if (idx >= BB * NN * DTOKN) return;
  int c = idx % 3;
  int t = idx / 3;
  int px = t & 7;  t >>= 3;
  int py = t & 7;  t >>= 3;
  int n = t % NN;  int b = t / NN;
  int gx = n & 15, gy = n >> 4;
  int row = gy * 8 + py, col = gx * 8 + px;
  p[idx] = img[(((long)b * 128 + row) * 128 + col) * 3 + c];
}

// ---------------- assemble h = [bos, text_emb, boi, img_e] -------------------
__global__ void embed_k(const int* __restrict__ ids, const float* __restrict__ emb,
                        const float* __restrict__ bos, const float* __restrict__ boi,
                        const float* __restrict__ imge, float* __restrict__ h)
{
  int idx = blockIdx.x * blockDim.x + threadIdx.x;
  if (idx >= BB * SS * DD) return;
  int d = idx % DD;
  int t = idx / DD;
  int s = t % SS, b = t / SS;
  float v;
  if (s == 0)           v = bos[d];
  else if (s <= TT)     v = emb[(long)ids[b * TT + (s - 1)] * DD + d];
  else if (s == TT + 1) v = boi[d];
  else                  v = imge[((long)b * NN + (s - TT - 2)) * DD + d];
  h[idx] = v;
}

// ---------------- split img-head raw -> mix_logits / mu / softplus(sc)+TOL ---
__global__ void split_k(const float* __restrict__ raw, float* __restrict__ out)
{
  const int NW = KMIX + 2 * KMIX * DAR;   // 32896
  long idx = (long)blockIdx.x * blockDim.x + threadIdx.x;
  long total = (long)BB * NN * NW;
  if (idx >= total) return;
  int  c = (int)(idx % NW);
  long rowi = idx / NW;                   // b*N + n
  float v = raw[idx];
  float* mix = out + (long)BB * TT * VV;
  float* mu  = mix + (long)BB * NN * KMIX;
  float* scp = mu  + (long)BB * NN * KMIX * DAR;
  if (c < KMIX)                  mix[rowi * KMIX + c] = v;
  else if (c < KMIX + KMIX*DAR)  mu[rowi * (KMIX * DAR) + (c - KMIX)] = v;
  else {
    float sp = (v > 20.f) ? v : log1pf(expf(v));
    scp[rowi * (KMIX * DAR) + (c - KMIX - KMIX * DAR)] = sp + 1e-6f;
  }
}

// -----------------------------------------------------------------------------
extern "C" void kernel_launch(void* const* d_in, const int* in_sizes, int n_in,
                              void* d_out, int out_size, void* d_ws, size_t ws_size,
                              hipStream_t stream)
{
  const int*   text_ids    = (const int*)  d_in[0];
  const float* images      = (const float*)d_in[1];
  const float* text_emb_w  = (const float*)d_in[2];
  const float* image_emb_w = (const float*)d_in[3];
  const float* image_emb_b = (const float*)d_in[4];
  const float* bos_emb     = (const float*)d_in[5];
  const float* boi_emb     = (const float*)d_in[6];
  const float* flow_w      = (const float*)d_in[7];
  const float* ln1_w       = (const float*)d_in[8];
  const float* wq          = (const float*)d_in[9];
  const float* wk          = (const float*)d_in[10];
  const float* wv          = (const float*)d_in[11];
  const float* wo          = (const float*)d_in[12];
  const float* ln2_w       = (const float*)d_in[13];
  const float* w1          = (const float*)d_in[14];
  const float* w2          = (const float*)d_in[15];
  const float* fn_w        = (const float*)d_in[16];
  const float* head_w      = (const float*)d_in[17];
  const float* head_b      = (const float*)d_in[18];
  float* out = (float*)d_out;
  float* ws  = (float*)d_ws;

  const int  ROWS = BB * SS;                 // 1028
  const int  NW   = KMIX + 2 * KMIX * DAR;   // 32896

  // workspace layout (floats) ~120 MB total
  long off = 0;
  float* h    = ws + off; off += (long)ROWS * DD;
  float* x    = ws + off; off += (long)ROWS * DD;
  float* q    = ws + off; off += (long)ROWS * DD;
  float* kb   = ws + off; off += (long)ROWS * HDIM;
  float* vb   = ws + off; off += (long)ROWS * HDIM;
  float* ao   = ws + off; off += (long)ROWS * DD;
  float* mid  = ws + off; off += (long)ROWS * FF;
  float* scr  = ws + off; off += (long)BB * HH * SS * SLD;
  float* pb   = ws + off; off += (long)BB * NN * DTOKN;
  float* lat  = ws + off; off += (long)BB * NN * DAR;
  float* ime  = ws + off; off += (long)BB * NN * DD;
  float* raw  = ws + off; off += (long)BB * NN * NW;

  auto cdiv = [](int a, int b) { return (a + b - 1) / b; };

  // 1. patchify
  { int tot = BB * NN * DTOKN;
    patchify_k<<<cdiv(tot, 256), 256, 0, stream>>>(images, pb); }

  // 2. lat = (p @ flow_w.T)[:, :128]  (flow_w stored (192,192); transB)
  gemm_wmma<<<dim3(cdiv(DAR,64), cdiv(BB*NN,128), 1), 256, 0, stream>>>(
      pb, DTOKN, 0, 0,  flow_w, DTOKN, 0, 0, 1,
      lat, DAR, 0, 0,  nullptr,  BB*NN, DAR, DTOKN, 1, 0, 0);

  // 3. img_e = lat @ image_emb_w + b
  gemm_wmma<<<dim3(cdiv(DD,64), cdiv(BB*NN,128), 1), 256, 0, stream>>>(
      lat, DAR, 0, 0,  image_emb_w, DD, 0, 0, 0,
      ime, DD, 0, 0,  image_emb_b,  BB*NN, DD, DAR, 1, 0, 0);

  // 4. assemble h
  { int tot = BB * SS * DD;
    embed_k<<<cdiv(tot, 256), 256, 0, stream>>>(text_ids, text_emb_w, bos_emb, boi_emb, ime, h); }

  // 5. transformer layers
  for (int l = 0; l < LL; ++l) {
    rmsnorm_k<<<ROWS, 256, 0, stream>>>(h, ln1_w + (long)l * DD, x);

    gemm_wmma<<<dim3(DD/64, cdiv(ROWS,128), 1), 256, 0, stream>>>(
        x, DD, 0, 0,  wq + (long)l*DD*DD, DD, 0, 0, 0,
        q, DD, 0, 0,  nullptr,  ROWS, DD, DD, 1, 0, 0);
    gemm_wmma<<<dim3(1, cdiv(ROWS,128), 1), 256, 0, stream>>>(
        x, DD, 0, 0,  wk + (long)l*DD*HDIM, HDIM, 0, 0, 0,
        kb, HDIM, 0, 0,  nullptr,  ROWS, HDIM, DD, 1, 0, 0);
    gemm_wmma<<<dim3(1, cdiv(ROWS,128), 1), 256, 0, stream>>>(
        x, DD, 0, 0,  wv + (long)l*DD*HDIM, HDIM, 0, 0, 0,
        vb, HDIM, 0, 0,  nullptr,  ROWS, HDIM, DD, 1, 0, 0);

    { int tot = BB*SS*HH*32; rope_k<<<cdiv(tot,256),256,0,stream>>>(q,  HH, 0.125f); }
    { int tot = BB*SS*1 *32; rope_k<<<cdiv(tot,256),256,0,stream>>>(kb, 1,  1.0f);   }

    // scores[b,h] = q[b,:,h,:] @ k[b]^T   (z = b*H + h), row stride SLD
    gemm_wmma<<<dim3(cdiv(SS,64), cdiv(SS,128), BB*HH), 256, 0, stream>>>(
        q, DD, (long)SS*DD, HDIM,  kb, HDIM, (long)SS*HDIM, 0, 1,
        scr, SLD, (long)HH*SS*SLD, (long)SS*SLD,  nullptr,
        SS, SS, HDIM, HH, 0, 0);

    softmax_k<<<BB*HH*SS, 256, 0, stream>>>(scr);

    // o[b,:,h,:] = probs[b,h] @ v[b]
    gemm_wmma<<<dim3(1, cdiv(SS,128), BB*HH), 256, 0, stream>>>(
        scr, SLD, (long)HH*SS*SLD, (long)SS*SLD,  vb, HDIM, (long)SS*HDIM, 0, 0,
        ao, DD, (long)SS*DD, HDIM,  nullptr,
        SS, HDIM, SS, HH, 0, 0);

    // h += o @ Wo
    gemm_wmma<<<dim3(DD/64, cdiv(ROWS,128), 1), 256, 0, stream>>>(
        ao, DD, 0, 0,  wo + (long)l*DD*DD, DD, 0, 0, 0,
        h, DD, 0, 0,  nullptr,  ROWS, DD, DD, 1, 1, 0);

    rmsnorm_k<<<ROWS, 256, 0, stream>>>(h, ln2_w + (long)l * DD, x);

    // mid = gelu(x @ W1)
    gemm_wmma<<<dim3(FF/64, cdiv(ROWS,128), 1), 256, 0, stream>>>(
        x, DD, 0, 0,  w1 + (long)l*DD*FF, FF, 0, 0, 0,
        mid, FF, 0, 0,  nullptr,  ROWS, FF, DD, 1, 0, 1);
    // h += mid @ W2
    gemm_wmma<<<dim3(DD/64, cdiv(ROWS,128), 1), 256, 0, stream>>>(
        mid, FF, 0, 0,  w2 + (long)l*FF*DD, DD, 0, 0, 0,
        h, DD, 0, 0,  nullptr,  ROWS, DD, FF, 1, 1, 0);
  }

  // 6. final norm
  rmsnorm_k<<<ROWS, 256, 0, stream>>>(h, fn_w, x);

  // 7. text_logits = hnorm[:, :T] @ text_emb_w^T   (z = batch)
  gemm_wmma<<<dim3(VV/64, cdiv(TT,128), BB), 256, 0, stream>>>(
      x, DD, (long)SS*DD, 0,  text_emb_w, DD, 0, 0, 1,
      out, VV, (long)TT*VV, 0,  nullptr,
      TT, VV, DD, 1, 0, 0);

  // 8. raw = hnorm[:, T+1:T+1+N] @ img_head_w + b
  gemm_wmma<<<dim3(cdiv(NW,64), cdiv(TT,128), BB), 256, 0, stream>>>(
      x + (long)(TT+1)*DD, DD, (long)SS*DD, 0,  head_w, NW, 0, 0, 0,
      raw, NW, (long)NN*NW, 0,  head_b,
      TT, NW, DD, 1, 0, 0);

  // 9. split into mix_logits / mu / softplus(sc)+TOL
  { long tot = (long)BB * NN * NW;
    split_k<<<(int)((tot + 255) / 256), 256, 0, stream>>>(raw, out); }
}
